// MatchingNetwork_80247168959298
// MI455X (gfx1250) — matrix operand order repaired and checked by reading, not compiled, not run on hardware
//
#include <hip/hip_runtime.h>
#include <hip/hip_bf16.h>
#include <math.h>

typedef __attribute__((ext_vector_type(16))) __bf16 v16bf;
typedef __attribute__((ext_vector_type(8)))  float  v8f;
typedef __attribute__((ext_vector_type(4)))  float  f32x4;

#define KDIM   20000
#define DDIM   64
#define SDIM   20
#define CDIM   20
#define BDIM   256
#define EPS_F  1e-10f

// ---------------------------------------------------------------------------
// Kernel 1: convert W [64, 20000] fp32 -> bf16 (stays hot in 192MB L2),
// and zero the two output accumulators (harness poisons d_out).
// ---------------------------------------------------------------------------
__global__ void convert_w_kernel(const float* __restrict__ W,
                                 __bf16* __restrict__ Wb,
                                 float* __restrict__ out2) {
    int i = blockIdx.x * blockDim.x + threadIdx.x;
    if (i < DDIM * KDIM) Wb[i] = (__bf16)W[i];
    if (blockIdx.x == 0 && threadIdx.x < 2) out2[threadIdx.x] = 0.0f;
}

// ---------------------------------------------------------------------------
// Kernel 2: GEMM  out[m, n] = sum_k X[m, k] * W[n, k] + bias[n]
//   X: [M, 20000] fp32 (streamed once, non-temporal)
//   W: [64, 20000] bf16 (L2-resident)
// Block = 256 threads = 8 waves. Wave w handles M-tile (w&1) and N-tile (w>>2? no: w>>1)
// of a 32-row x 64-col block tile. K-loop: 625 iters of v_wmma_f32_16x16x32_bf16.
//
// Per-lane register layouts used here (wave32; C/D as in CDNA5 ISA 7.12.2):
//   A 16x32 bf16: lane L: row m = L&15; lanes 0-15 hold K 0..15,
//                 lanes 16-31 hold K 16..31 (same order as B).
//   B 32x16 bf16: lane L: col n = L&15; lanes 0-15 hold K 0..15,
//                 lanes 16-31 hold K 16..31 (2 bf16 per VGPR, consecutive).
//   C/D 16x16 f32: lane L: col n = L&15; VGPR r -> row r + (L>=16 ? 8 : 0).
// ---------------------------------------------------------------------------
__global__ void gemm_wmma_kernel(const float* __restrict__ X,
                                 const __bf16* __restrict__ Wb,
                                 const float* __restrict__ bias,
                                 float* __restrict__ out) {
    const int wave = threadIdx.x >> 5;
    const int lane = threadIdx.x & 31;
    const int mt   = wave & 1;          // 2 M-tiles per block
    const int nt   = wave >> 1;         // 4 N-tiles (N = 64)
    const int mbase = blockIdx.x * 32 + mt * 16;
    const int nbase = nt * 16;
    const int l15   = lane & 15;
    const int kh    = lane >> 4;        // which K-half this lane holds

    // A: lane reads X[mbase+l15, k0 + kh*16 .. +15]
    const float*  arow = X  + (size_t)(mbase + l15) * KDIM + kh * 16;
    // B: lane reads Wb[nbase+l15, k0 + kh*16 .. +15]  (32 contiguous bytes)
    const __bf16* brow = Wb + (size_t)(nbase + l15) * KDIM + kh * 16;

    v8f acc = {};

#pragma unroll 2
    for (int k0 = 0; k0 < KDIM; k0 += 32) {
        // ---- A tile: 16 fp32 per lane, streamed non-temporally, cvt -> bf16
        f32x4 a0 = __builtin_nontemporal_load((const f32x4*)(arow + k0));
        f32x4 a1 = __builtin_nontemporal_load((const f32x4*)(arow + k0 + 4));
        f32x4 a2 = __builtin_nontemporal_load((const f32x4*)(arow + k0 + 8));
        f32x4 a3 = __builtin_nontemporal_load((const f32x4*)(arow + k0 + 12));
        v16bf a;
        a[0]  = (__bf16)a0.x; a[1]  = (__bf16)a0.y; a[2]  = (__bf16)a0.z; a[3]  = (__bf16)a0.w;
        a[4]  = (__bf16)a1.x; a[5]  = (__bf16)a1.y; a[6]  = (__bf16)a1.z; a[7]  = (__bf16)a1.w;
        a[8]  = (__bf16)a2.x; a[9]  = (__bf16)a2.y; a[10] = (__bf16)a2.z; a[11] = (__bf16)a2.w;
        a[12] = (__bf16)a3.x; a[13] = (__bf16)a3.y; a[14] = (__bf16)a3.z; a[15] = (__bf16)a3.w;

        // ---- B tile: 16 bf16 per lane (32B, aligned), already converted
        v16bf b = *(const v16bf*)(brow + k0);

        acc = __builtin_amdgcn_wmma_f32_16x16x32_bf16(
            /*neg_a=*/false, a, /*neg_b=*/false, b,
            /*c_mod=*/(short)0, acc, /*reuse_a=*/false, /*reuse_b=*/false);
    }

    // Epilogue: bias along N, scatter D per the C/D layout.
    const float bv   = bias[nbase + l15];
    const int   roff = kh * 8;
#pragma unroll
    for (int r = 0; r < 8; ++r) {
        int row = mbase + roff + r;
        out[(size_t)row * DDIM + nbase + l15] = acc[r] + bv;
    }
}

// ---------------------------------------------------------------------------
// Kernel 3: one wave per batch element. sims -> softmax -> preds -> argmax,
// log-softmax CE; atomically accumulate mean contributions into d_out[0..1].
// ---------------------------------------------------------------------------
__global__ void finalize_kernel(const float* __restrict__ sup,   // [B*S, 64]
                                const float* __restrict__ tgt,   // [B, 64]
                                const float* __restrict__ onehot,// [B, S, C]
                                const int*   __restrict__ ty,    // [B]
                                float* __restrict__ out2) {
    __shared__ float sims[SDIM];
    __shared__ float preds[CDIM];
    const int b    = blockIdx.x;
    const int lane = threadIdx.x;
    const float* tb = tgt + (size_t)b * DDIM;

    if (lane < SDIM) {
        const float* srow = sup + ((size_t)b * SDIM + lane) * DDIM;
        float nrm = 0.0f, dot = 0.0f;
#pragma unroll
        for (int d = 0; d < DDIM; ++d) {
            float s = srow[d];
            nrm += s * s;
            dot += s * tb[d];
        }
        sims[lane] = dot * rsqrtf(fmaxf(nrm, EPS_F));
    }
    __syncthreads();

    // softmax over the 20 sims (computed redundantly per lane; trivial)
    float m = -INFINITY;
    for (int s = 0; s < SDIM; ++s) m = fmaxf(m, sims[s]);
    float den = 0.0f;
    for (int s = 0; s < SDIM; ++s) den += expf(sims[s] - m);

    if (lane < CDIM) {
        const float* oh = onehot + ((size_t)b * SDIM) * CDIM + lane;
        float p = 0.0f;
        for (int s = 0; s < SDIM; ++s) p += expf(sims[s] - m) * oh[s * CDIM];
        preds[lane] = p / den;
    }
    __syncthreads();

    if (lane == 0) {
        const int t = ty[b];
        float pm = preds[0];
        int   idx = 0;
        for (int c = 1; c < CDIM; ++c)
            if (preds[c] > pm) { pm = preds[c]; idx = c; }   // first-max like jnp.argmax
        float s = 0.0f;
        for (int c = 0; c < CDIM; ++c) s += expf(preds[c] - pm);
        float logp = preds[t] - pm - logf(s);
        atomicAdd(&out2[0], (idx == t) ? (1.0f / (float)BDIM) : 0.0f);
        atomicAdd(&out2[1], -logp / (float)BDIM);
    }
}

// ---------------------------------------------------------------------------
extern "C" void kernel_launch(void* const* d_in, const int* in_sizes, int n_in,
                              void* d_out, int out_size, void* d_ws, size_t ws_size,
                              hipStream_t stream) {
    (void)in_sizes; (void)n_in; (void)out_size; (void)ws_size;

    const float* supimg = (const float*)d_in[0];   // [256, 20, 20000]
    const float* onehot = (const float*)d_in[1];   // [256, 20, 20]
    const float* timg   = (const float*)d_in[2];   // [256, 20000]
    const int*   ty     = (const int*)  d_in[3];   // [256]
    const float* W      = (const float*)d_in[4];   // [64, 20000]
    const float* bias   = (const float*)d_in[5];   // [64]
    float* out = (float*)d_out;                    // [accuracy, loss]

    char* ws = (char*)d_ws;
    __bf16* Wb  = (__bf16*)ws;                                    // 2,560,000 B
    float*  sup = (float*)(ws + 2560000);                         // 1,310,720 B
    float*  tgt = (float*)(ws + 2560000 + 1310720);               //    65,536 B

    // 1) W -> bf16 (L2-resident), zero accumulators
    convert_w_kernel<<<(DDIM * KDIM + 255) / 256, 256, 0, stream>>>(W, Wb, out);

    // 2) support encodings: M = 5120 rows -> 160 blocks of 32 rows
    gemm_wmma_kernel<<<160, 256, 0, stream>>>(supimg, Wb, bias, sup);

    // 3) target encodings: M = 256 rows -> 8 blocks
    gemm_wmma_kernel<<<8, 256, 0, stream>>>(timg, Wb, bias, tgt);

    // 4) similarities / softmax / classify / reduce
    finalize_kernel<<<BDIM, 32, 0, stream>>>(sup, tgt, onehot, ty, out);
}
